// BioLinearAttention_1778116460844
// MI455X (gfx1250) — matrix operand order repaired and checked by reading, not compiled, not run on hardware
//
#include <hip/hip_runtime.h>
#include <hip/hip_bf16.h>

typedef __bf16 bf16;
typedef __attribute__((ext_vector_type(8)))  __bf16 v8bf;
typedef __attribute__((ext_vector_type(16))) __bf16 v16bf;
typedef __attribute__((ext_vector_type(8)))  float  v8f;
typedef __attribute__((ext_vector_type(4))) unsigned int u32x4;
typedef __attribute__((ext_vector_type(8))) int          i32x8;
typedef __attribute__((ext_vector_type(4))) int          i32x4;

#define WMMA_BF16(a, b, c) \
  __builtin_amdgcn_wmma_f32_16x16x32_bf16(false, (a), false, (b), (short)0, (c), false, false)

static constexpr int Bb  = 4;
static constexpr int Tt  = 4096;
static constexpr int Cc  = 1024;
static constexpr int Hh  = 16;
static constexpr int Dd  = 64;
static constexpr int Mtot = Bb * Tt;       // 16384
static constexpr int KSPLIT = 8;

// LDS B-tile geometry for proj_gemm: 128 N-rows x 64 K-elems, padded row stride
static constexpr int KC     = 64;   // K-chunk staged per barrier
static constexpr int BN     = 128;  // N rows per block tile
static constexpr int LDSROW = 72;   // 64 elems + 8-elem (16B) pad per 128B -> bank-conflict-free

#if __has_builtin(__builtin_amdgcn_tensor_load_to_lds) && \
    __has_builtin(__builtin_amdgcn_s_wait_tensorcnt)
#define USE_TDM 1
#else
#define USE_TDM 0
#endif

__device__ __forceinline__ float elu1(float f) {
  return f > 0.f ? f + 1.f : __expf(f);
}

__device__ __forceinline__ v16bf cat8(v8bf lo, v8bf hi) {
  return __builtin_shufflevector(lo, hi, 0,1,2,3,4,5,6,7,8,9,10,11,12,13,14,15);
}

#if USE_TDM
// Issue a TDM 2D-tile load: KC x BN tile of Bt (row stride K elems, 2B elems)
// into LDS at lds_addr, with 16B padding per 128B row (LDS row stride = 144B).
__device__ __forceinline__ void tdm_load_b_tile(unsigned lds_addr,
                                                const bf16* gsrc, int K, int N) {
  union { unsigned u[4]; u32x4 v; } g0;
  union { unsigned u[8]; i32x8 v; } g1;
  union { unsigned u[4]; i32x4 v; } gz;
  unsigned long long ga = (unsigned long long)(size_t)gsrc;
  g0.u[0] = 1u;                                   // count=1, user mode
  g0.u[1] = lds_addr;                             // LDS byte address
  g0.u[2] = (unsigned)ga;                         // global_addr[31:0]
  g0.u[3] = (unsigned)(ga >> 32) | (2u << 30);    // global_addr[56:32] | type=2
  // group1: data_size=1(2B) bit16; pad_enable bit20; pad_interval[24:22]=4 (32dw=128B);
  // pad_amount[31:25]=3 (4dw=16B)
  g1.u[0] = (1u << 16) | (1u << 20) | (4u << 22) | (3u << 25);
  g1.u[1] = ((unsigned)K & 0xFFFFu) << 16;                         // tensor_dim0[15:0]
  g1.u[2] = (((unsigned)K >> 16) & 0xFFFFu) |
            (((unsigned)N & 0xFFFFu) << 16);                       // dim0 hi | dim1 lo
  g1.u[3] = (((unsigned)N >> 16) & 0xFFFFu) | ((unsigned)KC << 16); // dim1 hi | tile_dim0
  g1.u[4] = (unsigned)BN;                                          // tile_dim1 | tile_dim2=0
  g1.u[5] = (unsigned)K;                                           // tensor_dim0_stride lo32
  g1.u[6] = 0u;                                                    // stride hi | dim1_stride lo
  g1.u[7] = 0u;
  gz.u[0] = 0u; gz.u[1] = 0u; gz.u[2] = 0u; gz.u[3] = 0u;
#if defined(__clang_major__) && (__clang_major__ >= 23)
  union { unsigned u[8]; i32x8 v; } gz8;
  for (int i = 0; i < 8; ++i) gz8.u[i] = 0u;
  __builtin_amdgcn_tensor_load_to_lds(g0.v, g1.v, gz.v, gz.v, gz8.v, 0);
#else
  __builtin_amdgcn_tensor_load_to_lds(g0.v, g1.v, gz.v, gz.v, 0);
#endif
}
#endif

// ---------------------------------------------------------------------------
// fp32 -> bf16 conversion
// ---------------------------------------------------------------------------
__global__ __launch_bounds__(256) void cvt_f32_bf16(const float* __restrict__ src,
                                                    bf16* __restrict__ dst, int n) {
  int i = (blockIdx.x * blockDim.x + threadIdx.x) * 4;
  int stride = gridDim.x * blockDim.x * 4;
  for (; i < n; i += stride) {
    float4 v = *(const float4*)(src + i);
    dst[i + 0] = (bf16)v.x; dst[i + 1] = (bf16)v.y;
    dst[i + 2] = (bf16)v.z; dst[i + 3] = (bf16)v.w;
  }
}

// ---------------------------------------------------------------------------
// Tiled GEMM: C = A[M,K] @ Bt[N,K]^T (+bias).
// Block: 8 waves as 4(M) x 2(N); wave tile 32x64 (8 acc frags).
// Block tile 128(M) x 128(N). B tile staged in LDS via TDM, double-buffered.
// ---------------------------------------------------------------------------
enum { EP_Q = 0, EP_KT = 1, EP_VT = 2, EP_OUT = 3 };

template <int MODE>
__global__ __launch_bounds__(256)
void proj_gemm(const bf16* __restrict__ A, const bf16* __restrict__ Bt,
               const float* __restrict__ bias, void* __restrict__ Cout,
               int M, int N, int K) {
  __shared__ __align__(256) bf16 Bs[2][BN * LDSROW];   // 2 x 18KB

  const int lane = threadIdx.x & 31;
  const int wave = threadIdx.x >> 5;
  const int r    = lane & 15;
  const int hi   = lane >> 4;
  const int mw   = wave >> 1;                 // 0..3
  const int nw   = wave & 1;                  // 0..1
  const int m0   = blockIdx.x * 128 + mw * 32;
  const int n0b  = blockIdx.y * BN;           // block N origin

  const bf16* a0 = A + (size_t)(m0 + r) * K;
  const bf16* a1 = a0 + (size_t)16 * K;

  v8f acc0[4] = {};
  v8f acc1[4] = {};

  const int nchunks = K / KC;

  // stage chunk 0
#if USE_TDM
  if (wave == 0) {
    tdm_load_b_tile((unsigned)(size_t)(void*)&Bs[0][0],
                    Bt + (size_t)n0b * K, K, N);
    __builtin_amdgcn_s_wait_tensorcnt(0);
  }
#else
  for (int cc = threadIdx.x; cc < BN * (KC / 8); cc += 256) {
    int row = cc >> 3, kk = (cc & 7) * 8;
    *(v8bf*)&Bs[0][row * LDSROW + kk] = *(const v8bf*)(Bt + (size_t)(n0b + row) * K + kk);
  }
#endif
  __syncthreads();

  for (int c = 0; c < nchunks; ++c) {
    const int kc = c * KC;
    // prefetch next chunk into the other buffer
    if (c + 1 < nchunks) {
#if USE_TDM
      if (wave == 0)
        tdm_load_b_tile((unsigned)(size_t)(void*)&Bs[(c + 1) & 1][0],
                        Bt + (size_t)n0b * K + (kc + KC), K, N);
#else
      for (int cc = threadIdx.x; cc < BN * (KC / 8); cc += 256) {
        int row = cc >> 3, kk = (cc & 7) * 8;
        *(v8bf*)&Bs[(c + 1) & 1][row * LDSROW + kk] =
            *(const v8bf*)(Bt + (size_t)(n0b + row) * K + (kc + KC) + kk);
      }
#endif
    }

    const bf16* bs = &Bs[c & 1][0];
#pragma unroll
    for (int ks = 0; ks < KC; ks += 32) {
      const int ka = kc + ks + hi * 8;
      v16bf aA = cat8(*(const v8bf*)(a0 + ka), *(const v8bf*)(a0 + ka + 16));
      v16bf aB = cat8(*(const v8bf*)(a1 + ka), *(const v8bf*)(a1 + ka + 16));
      if (ks == 0 && kc + KC < K) {
        __builtin_prefetch(a0 + kc + KC, 0, 3);
        __builtin_prefetch(a1 + kc + KC, 0, 3);
      }
      const int kk = ks + hi * 8;
#pragma unroll
      for (int j = 0; j < 4; ++j) {
        const bf16* bp = bs + (nw * 64 + j * 16 + r) * LDSROW + kk;
        v16bf bfr = cat8(*(const v8bf*)bp, *(const v8bf*)(bp + 16));
        acc0[j] = WMMA_BF16(aA, bfr, acc0[j]);
        acc1[j] = WMMA_BF16(aB, bfr, acc1[j]);
      }
    }

#if USE_TDM
    if (wave == 0 && c + 1 < nchunks) __builtin_amdgcn_s_wait_tensorcnt(0);
#endif
    __syncthreads();
  }

  // ---- epilogue: two 16-row groups ----
#pragma unroll
  for (int g = 0; g < 2; ++g) {
    const v8f* acc = g ? acc1 : acc0;
    const int mb = m0 + g * 16 + 8 * hi;
#pragma unroll
    for (int j = 0; j < 4; ++j) {
      const int col = n0b + nw * 64 + j * 16 + r;
      const float bv = bias[col];
      if (MODE == EP_Q) {
        bf16* out = (bf16*)Cout;
#pragma unroll
        for (int e = 0; e < 8; ++e) {
          float f = acc[j][e] + bv;
          out[(size_t)(mb + e) * N + col] = (bf16)elu1(f);
        }
      } else if (MODE == EP_KT || MODE == EP_VT) {
        const int b_ = mb >> 12;
        const int t  = mb & (Tt - 1);
        const int h  = col >> 6;
        const int d  = col & (Dd - 1);
        bf16* dst = (bf16*)Cout + ((size_t)(b_ * Hh + h) * Dd + d) * Tt + t;
#pragma unroll
        for (int e = 0; e < 8; ++e) {
          float f = acc[j][e] + bv;
          if (MODE == EP_KT) f = elu1(f);
          dst[e] = (bf16)f;
        }
      } else {  // EP_OUT
        float* out = (float*)Cout;
#pragma unroll
        for (int e = 0; e < 8; ++e)
          out[(size_t)(mb + e) * N + col] = acc[j][e] + bv;
      }
    }
  }
}

// ---------------------------------------------------------------------------
// kv^T[e,d] = sum_t v[t,e] * k[t,d]   (per (b,h); split-K over T, partials)
// ---------------------------------------------------------------------------
__global__ __launch_bounds__(128)
void kv_gemm(const bf16* __restrict__ vt, const bf16* __restrict__ kt,
             float* __restrict__ part) {
  const int s    = blockIdx.x;
  const int bh   = blockIdx.y;
  const int lane = threadIdx.x & 31;
  const int wave = threadIdx.x >> 5;
  const int r    = lane & 15;
  const int hi   = lane >> 4;
  const int m0   = wave * 16;

  const bf16* Abase = vt + (size_t)bh * Dd * Tt;
  const bf16* Bt    = kt + (size_t)bh * Dd * Tt;
  const bf16* arow  = Abase + (size_t)(m0 + r) * Tt;

  v8f acc[4] = {};
  const int kchunk = Tt / KSPLIT;
  const int kbeg = s * kchunk;
  for (int k0 = kbeg; k0 < kbeg + kchunk; k0 += 32) {
    const int ka = k0 + hi * 8;
    v16bf a = cat8(*(const v8bf*)(arow + ka), *(const v8bf*)(arow + ka + 16));
    __builtin_prefetch(arow + ka + 32, 0, 3);
#pragma unroll
    for (int j = 0; j < 4; ++j) {
      v16bf bfr = cat8(*(const v8bf*)(Bt + (size_t)(j * 16 + r) * Tt + k0 + hi * 8), *(const v8bf*)(Bt + (size_t)(j * 16 + r) * Tt + k0 + hi * 8 + 16));
      acc[j] = WMMA_BF16(a, bfr, acc[j]);
    }
  }

  float* dst = part + (size_t)(s * (Bb * Hh) + bh) * (Dd * Dd);
  const int mb = m0 + 8 * hi;
#pragma unroll
  for (int j = 0; j < 4; ++j) {
    const int col = j * 16 + r;
#pragma unroll
    for (int e = 0; e < 8; ++e)
      dst[(mb + e) * Dd + col] = acc[j][e];
  }
}

__global__ __launch_bounds__(256)
void kv_reduce(const float* __restrict__ part, bf16* __restrict__ kvbf) {
  const int i = blockIdx.x * blockDim.x + threadIdx.x;
  const int bh = i >> 12;
  const int ed = i & 4095;
  float s = 0.f;
#pragma unroll
  for (int k = 0; k < KSPLIT; ++k)
    s += part[(size_t)(k * (Bb * Hh) + bh) * 4096 + ed];
  kvbf[i] = (bf16)s;
}

__global__ __launch_bounds__(256)
void ksum_ker(const bf16* __restrict__ kt, float* __restrict__ ksum) {
  const int row  = blockIdx.x * 8 + (threadIdx.x >> 5);
  const int lane = threadIdx.x & 31;
  const bf16* p = kt + (size_t)row * Tt;
  float s = 0.f;
  for (int t0 = lane * 8; t0 < Tt; t0 += 32 * 8) {
    v8bf v = *(const v8bf*)(p + t0);
#pragma unroll
    for (int i = 0; i < 8; ++i) s += (float)v[i];
  }
#pragma unroll
  for (int off = 16; off > 0; off >>= 1) s += __shfl_xor(s, off, 32);
  if (lane == 0) ksum[row] = s;
}

__global__ __launch_bounds__(256)
void z_ker(const bf16* __restrict__ qb, const float* __restrict__ ksum,
           float* __restrict__ z) {
  const int idx = blockIdx.x * blockDim.x + threadIdx.x;
  const int t  = idx & (Tt - 1);
  const int bh = idx >> 12;
  const int b  = bh >> 4;
  const int h  = bh & 15;
  const bf16* qrow = qb + ((size_t)(b * Tt) + t) * Cc + h * Dd;
  const float* ks  = ksum + bh * Dd;
  float s = 0.f;
#pragma unroll 8
  for (int d = 0; d < Dd; ++d) s += (float)qrow[d] * ks[d];
  z[idx] = 1.f / (s + 1e-6f);
}

__global__ __launch_bounds__(256)
void y_gemm(const bf16* __restrict__ qb, const bf16* __restrict__ kvbf,
            const float* __restrict__ z, bf16* __restrict__ yb) {
  const int bh = blockIdx.y;
  const int b  = bh >> 4;
  const int h  = bh & 15;
  const int lane = threadIdx.x & 31;
  const int wave = threadIdx.x >> 5;
  const int r  = lane & 15;
  const int hi = lane >> 4;
  const int m0 = blockIdx.x * 128 + wave * 16;

  const bf16* arow = qb + ((size_t)(b * Tt) + m0 + r) * Cc + h * Dd;
  const bf16* Bt   = kvbf + (size_t)bh * (Dd * Dd);

  v8f acc[4] = {};
#pragma unroll
  for (int k0 = 0; k0 < Dd; k0 += 32) {
    const int ka = k0 + hi * 8;
    v16bf a = cat8(*(const v8bf*)(arow + ka), *(const v8bf*)(arow + ka + 16));
#pragma unroll
    for (int j = 0; j < 4; ++j) {
      v16bf bfr = cat8(*(const v8bf*)(Bt + (j * 16 + r) * Dd + k0 + hi * 8), *(const v8bf*)(Bt + (j * 16 + r) * Dd + k0 + hi * 8 + 16));
      acc[j] = WMMA_BF16(a, bfr, acc[j]);
    }
  }

  const int mb = m0 + 8 * hi;
#pragma unroll
  for (int j = 0; j < 4; ++j) {
    const int col = j * 16 + r;
#pragma unroll
    for (int e = 0; e < 8; ++e) {
      const float zz = z[(size_t)bh * Tt + mb + e];
      yb[((size_t)(b * Tt) + mb + e) * Cc + h * Dd + col] = (bf16)(acc[j][e] * zz);
    }
  }
}

// ---------------------------------------------------------------------------
extern "C" void kernel_launch(void* const* d_in, const int* in_sizes, int n_in,
                              void* d_out, int out_size, void* d_ws, size_t ws_size,
                              hipStream_t stream) {
  (void)in_sizes; (void)n_in; (void)out_size; (void)ws_size;

  const float* x  = (const float*)d_in[0];
  const float* Wq = (const float*)d_in[1];
  const float* bq = (const float*)d_in[2];
  const float* Wk = (const float*)d_in[3];
  const float* bk = (const float*)d_in[4];
  const float* Wv = (const float*)d_in[5];
  const float* bv = (const float*)d_in[6];
  const float* Wc = (const float*)d_in[7];
  const float* bc = (const float*)d_in[8];

  char* ws = (char*)d_ws;
  size_t off = 0;
  auto carve = [&](size_t bytes) {
    void* p = ws + off;
    off = (off + bytes + 255) & ~(size_t)255;
    return p;
  };

  bf16* wqb  = (bf16*)carve((size_t)Cc * Cc * 2);
  bf16* wkb  = (bf16*)carve((size_t)Cc * Cc * 2);
  bf16* wvb  = (bf16*)carve((size_t)Cc * Cc * 2);
  bf16* wcb  = (bf16*)carve((size_t)Cc * Cc * 2);
  bf16* xb   = (bf16*)carve((size_t)Mtot * Cc * 2);
  bf16* qbuf = (bf16*)carve((size_t)Mtot * Cc * 2);
  bf16* ktb  = (bf16*)carve((size_t)Bb * Hh * Dd * Tt * 2);
  bf16* vtb  = (bf16*)carve((size_t)Bb * Hh * Dd * Tt * 2);
  bf16* ybuf = (bf16*)carve((size_t)Mtot * Cc * 2);
  float* part = (float*)carve((size_t)KSPLIT * Bb * Hh * Dd * Dd * 4);
  bf16* kvbf  = (bf16*)carve((size_t)Bb * Hh * Dd * Dd * 2);
  float* ksum = (float*)carve((size_t)Bb * Hh * Dd * 4);
  float* zbuf = (float*)carve((size_t)Bb * Hh * Tt * 4);

  // 1) inputs -> bf16
  cvt_f32_bf16<<<1024, 256, 0, stream>>>(Wq, wqb, Cc * Cc);
  cvt_f32_bf16<<<1024, 256, 0, stream>>>(Wk, wkb, Cc * Cc);
  cvt_f32_bf16<<<1024, 256, 0, stream>>>(Wv, wvb, Cc * Cc);
  cvt_f32_bf16<<<1024, 256, 0, stream>>>(Wc, wcb, Cc * Cc);
  cvt_f32_bf16<<<8192, 256, 0, stream>>>(x, xb, Mtot * Cc);

  // 2) projections (bf16 WMMA, TDM-staged B tiles)
  dim3 pgrid(Mtot / 128, Cc / BN);
  proj_gemm<EP_Q ><<<pgrid, 256, 0, stream>>>(xb, wqb, bq, qbuf, Mtot, Cc, Cc);
  proj_gemm<EP_KT><<<pgrid, 256, 0, stream>>>(xb, wkb, bk, ktb,  Mtot, Cc, Cc);
  proj_gemm<EP_VT><<<pgrid, 256, 0, stream>>>(xb, wvb, bv, vtb,  Mtot, Cc, Cc);

  // 3) k_sum, kv = k^T v (split-K, deterministic partial reduce)
  ksum_ker<<<(Bb * Hh * Dd) / 8, 256, 0, stream>>>(ktb, ksum);
  kv_gemm<<<dim3(KSPLIT, Bb * Hh), 128, 0, stream>>>(vtb, ktb, part);
  kv_reduce<<<(Bb * Hh * Dd * Dd) / 256, 256, 0, stream>>>(part, kvbf);

  // 4) normalizer z, y = (q @ kv) * z
  z_ker<<<(Bb * Hh * Tt) / 256, 256, 0, stream>>>(qbuf, ksum, zbuf);
  y_gemm<<<dim3(Tt / 128, Bb * Hh), 256, 0, stream>>>(qbuf, kvbf, zbuf, ybuf);

  // 5) output projection, fp32 + bias into d_out
  proj_gemm<EP_OUT><<<pgrid, 256, 0, stream>>>(ybuf, wcb, bc, d_out, Mtot, Cc, Cc);
}
